// NNFMLoss_44813688766518
// MI455X (gfx1250) — matrix operand order repaired and checked by reading, not compiled, not run on hardware
//
#include <hip/hip_runtime.h>
#include <hip/hip_bf16.h>

typedef __attribute__((ext_vector_type(16))) _Float16 v16h;
typedef __attribute__((ext_vector_type(8)))  float    v8f;
typedef unsigned int u32x4 __attribute__((ext_vector_type(4)));
typedef unsigned int u32x8 __attribute__((ext_vector_type(8)));

#define CCH   768
#define HWN   12544     // 112*112
#define EPSF  1e-8f

// ---------------------------------------------------------------------------
// Pass 1: per-column sum of squares + transpose/convert to K-contiguous f16.
// ---------------------------------------------------------------------------
__global__ void colstats_kernel(const float* __restrict__ a,
                                const float* __restrict__ b,
                                float* __restrict__ ssa, float* __restrict__ ssb,
                                _Float16* __restrict__ a16, _Float16* __restrict__ b16) {
  int i = blockIdx.x * blockDim.x + threadIdx.x;
  if (i >= HWN) return;
  float sa = 0.f, sb = 0.f;
  for (int c = 0; c < CCH; ++c) {
    float av = a[(size_t)c * HWN + i];
    float bv = b[(size_t)c * HWN + i];
    sa += av * av;
    sb += bv * bv;
    a16[(size_t)i * CCH + c] = (_Float16)av;
  }
  ssa[i] = sa;
  ssb[i] = sb;
  float invb = 1.0f / (sqrtf(sb + EPSF) + EPSF);
  for (int c = 0; c < CCH; ++c)
    b16[(size_t)i * CCH + c] = (_Float16)(b[(size_t)c * HWN + i] * invb);
}

__global__ void init_kernel(unsigned long long* __restrict__ best, float* __restrict__ out) {
  int i = blockIdx.x * blockDim.x + threadIdx.x;
  if (i < HWN) best[i] = 0ull;
  if (i == 0) out[0] = 0.0f;
}

__device__ __forceinline__ unsigned int fkey(float f) {
  unsigned int u = __float_as_uint(f);
  return (u & 0x80000000u) ? ~u : (u | 0x80000000u);   // monotone float->uint
}

// ---------------------------------------------------------------------------
// TDM: issue a 2D tile load (elements are 2 bytes) into LDS.
// D# built per CDNA5 ISA.
// ---------------------------------------------------------------------------
__device__ __forceinline__ void tdm_load_2d(unsigned lds_off, const void* gptr,
                                            unsigned tile_d0, unsigned tile_d1,
                                            unsigned tensor_d0, unsigned tensor_d1,
                                            unsigned stride_d0) {
  unsigned long long ga = (unsigned long long)gptr;
  u32x4 g0 = {
      1u,                                   // count=1, user descriptor, no gather
      lds_off,                              // lds_addr (bytes, wave-LDS relative)
      (unsigned)(ga & 0xFFFFFFFFu),         // global_addr[31:0]
      ((unsigned)((ga >> 32) & 0x01FFFFFFu)) | 0x80000000u  // ga[56:32] | type=2
  };
  u32x8 g1 = {
      0x00010000u,                                          // data_size=1 (2B), mask=0
      (tensor_d0 & 0xFFFFu) << 16,                          // dim0[15:0] -> bits 63:48
      ((tensor_d0 >> 16) & 0xFFFFu) | ((tensor_d1 & 0xFFFFu) << 16),
      ((tensor_d1 >> 16) & 0xFFFFu) | ((tile_d0 & 0xFFFFu) << 16),
      (tile_d1 & 0xFFFFu),                                  // tile_dim1; tile_dim2=0
      stride_d0,                                            // dim0_stride[31:0]
      0u,                                                   // dim0_stride hi / dim1_stride lo
      0u
  };
  asm volatile("tensor_load_to_lds %0, %1" :: "s"(g0), "s"(g1) : "memory");
}

// ---------------------------------------------------------------------------
// Pass 2: S[i,j] = sum_c a16[i][c]*b16[j][c], fused row-argmax.
// Block tile 128(M) x 64(N); K in steps of 32 (v_wmma_f32_16x16x32_f16);
// TDM-staged, double-buffered LDS; 8 waves, wave w owns rows [w*16, w*16+16).
// ---------------------------------------------------------------------------
__launch_bounds__(256)
__global__ void gemm_argmax_kernel(const _Float16* __restrict__ a16,
                                   const _Float16* __restrict__ b16,
                                   unsigned long long* __restrict__ best) {
  // buffer p: A tile at p*6144 (128x32 halves), B tile at p*6144+4096 (64x32 halves)
  __shared__ __align__(16) _Float16 smem[2 * 6144];

  const int tid  = threadIdx.x;
  const int lane = tid & 31;
  const int w    = tid >> 5;
  const int i0   = blockIdx.x * 128;
  const int j0   = blockIdx.y * 64;
  const unsigned ldsbase = (unsigned)(unsigned long long)(void*)&smem[0];

  // Prologue: stage K-slab 0 into buffer 0.
  if (w == 0) {
    tdm_load_2d(ldsbase, a16 + (size_t)i0 * CCH, 32u, 128u, CCH, HWN, CCH);
    tdm_load_2d(ldsbase + 4096u * 2u, b16 + (size_t)j0 * CCH, 32u, 64u, CCH, HWN, CCH);
    __builtin_amdgcn_s_wait_tensorcnt(0);
  }
  __syncthreads();

  v8f acc[4] = {v8f{}, v8f{}, v8f{}, v8f{}};

  for (int kt = 0; kt < 24; ++kt) {
    const int p = kt & 1;
    const _Float16* As = smem + p * 6144;
    const _Float16* Bs = As + 4096;

    // Kick off next K-slab into the other buffer (overlaps with WMMA below).
    if (w == 0 && kt + 1 < 24) {
      const int k1 = (kt + 1) * 32;
      const unsigned pb = ldsbase + (unsigned)((p ^ 1) * 6144 * 2);
      tdm_load_2d(pb, a16 + (size_t)i0 * CCH + k1, 32u, 128u, CCH, HWN, CCH);
      tdm_load_2d(pb + 4096u * 2u, b16 + (size_t)j0 * CCH + k1, 32u, 64u, CCH, HWN, CCH);
    }

    // A fragment (16x32 f16): lane L<16 -> row L, K {0..7,16..23}; L>=16 -> K {8..15,24..31}
    v16h af;
    {
      int r = lane & 15, kh = lane >> 4;
      ((uint4*)&af)[0] = *(const uint4*)(&As[(w * 16 + r) * 32 + kh * 8]);
      ((uint4*)&af)[1] = *(const uint4*)(&As[(w * 16 + r) * 32 + 16 + kh * 8]);
    }
#pragma unroll
    for (int nt = 0; nt < 4; ++nt) {
      // B fragment (32x16 f16): lane L -> col L&15, K {0..7,16..23} (L<16) or {8..15,24..31} (L>=16)
      v16h bf;
      {
        int c = lane & 15, kb = (lane >> 4) * 8;
        const _Float16* pB = &Bs[(nt * 16 + c) * 32 + kb];
        ((uint4*)&bf)[0] = *(const uint4*)(pB);
        ((uint4*)&bf)[1] = *(const uint4*)(pB + 16);
      }
      acc[nt] = __builtin_amdgcn_wmma_f32_16x16x32_f16(
          /*neg_a=*/false, af, /*neg_b=*/false, bf,
          /*c_mod=*/(short)0, acc[nt], /*reuse_a=*/false, /*reuse_b=*/false);
    }

    // DMA for slab kt+1 must be complete before anyone reads it next iteration.
    if (w == 0 && kt + 1 < 24) __builtin_amdgcn_s_wait_tensorcnt(0);
    __syncthreads();
  }

  // Epilogue: C/D layout -> VGPR v, lanes 0-15: M=v, N=lane; lanes 16-31: M=v+8, N=lane-16
  const int hi    = lane >> 4;
  const int l15   = lane & 15;
  const int mbase = i0 + w * 16 + hi * 8;
#pragma unroll
  for (int v = 0; v < 8; ++v) {
    unsigned long long pk = 0ull;
#pragma unroll
    for (int nt = 0; nt < 4; ++nt) {
      float val = acc[nt][v];
      unsigned int j = (unsigned int)(j0 + nt * 16 + l15);
      unsigned long long e =
          ((unsigned long long)fkey(val) << 32) | (unsigned long long)(~j);
      pk = (e > pk) ? e : pk;  // ~j: ties resolve to smallest j (argmin semantics)
    }
#pragma unroll
    for (int off = 1; off < 16; off <<= 1) {
      unsigned long long o = (unsigned long long)__shfl_xor((unsigned long long)pk, off, 16);
      pk = (o > pk) ? o : pk;
    }
    if (l15 == 0) atomicMax(&best[mbase + v], pk);
  }
}

// ---------------------------------------------------------------------------
// Pass 3: exact fp32 cosine for the selected pair, mean(1 - cos).
// ---------------------------------------------------------------------------
__global__ void loss_kernel(const float* __restrict__ a, const float* __restrict__ b,
                            const float* __restrict__ ssa, const float* __restrict__ ssb,
                            const unsigned long long* __restrict__ best,
                            float* __restrict__ out) {
  int i = blockIdx.x * blockDim.x + threadIdx.x;
  if (i >= HWN) return;
  unsigned int j = ~(unsigned int)(best[i] & 0xFFFFFFFFull);
  float dot = 0.f;
  for (int c = 0; c < CCH; ++c)
    dot += a[(size_t)c * HWN + i] * b[(size_t)c * HWN + j];
  float cs = dot / (sqrtf(ssa[i]) + EPSF) / (sqrtf(ssb[j]) + EPSF);
  atomicAdd(out, (1.0f - cs) * (1.0f / (float)HWN));
}

extern "C" void kernel_launch(void* const* d_in, const int* in_sizes, int n_in,
                              void* d_out, int out_size, void* d_ws, size_t ws_size,
                              hipStream_t stream) {
  const float* a = (const float*)d_in[0];  // x_feats
  const float* b = (const float*)d_in[1];  // s_feats
  float* out = (float*)d_out;

  char* ws = (char*)d_ws;
  float* ssa = (float*)ws;                                  // HWN f32
  float* ssb = ssa + HWN;                                   // HWN f32
  unsigned long long* best = (unsigned long long*)(ws + (size_t)2 * HWN * 4);  // HWN u64
  _Float16* a16 = (_Float16*)(ws + (size_t)2 * HWN * 4 + (size_t)HWN * 8);     // HWN*C f16
  _Float16* b16 = a16 + (size_t)HWN * CCH;                                     // HWN*C f16

  const int tb = 256;
  const int nb = (HWN + tb - 1) / tb;  // 49

  init_kernel<<<nb, tb, 0, stream>>>(best, out);
  colstats_kernel<<<nb, tb, 0, stream>>>(a, b, ssa, ssb, a16, b16);
  dim3 grid(HWN / 128, HWN / 64);      // 98 x 196
  gemm_argmax_kernel<<<grid, tb, 0, stream>>>(a16, b16, best);
  loss_kernel<<<nb, tb, 0, stream>>>(a, b, ssa, ssb, best, out);
}
